// DeformableProjEmbed_94489281106
// MI455X (gfx1250) — matrix operand order repaired and checked by reading, not compiled, not run on hardware
//
#include <hip/hip_runtime.h>
#include <math.h>

// ---------------------------------------------------------------------------
// DeformableProjEmbed for MI455X (gfx1250, wave32, WMMA, TDM double-buffer).
//  x[8,512,64,64] f32 -> out[8,128,64,64] f32
//  proj_b is skipped on purpose: BN (batch stats) cancels per-channel biases.
// ---------------------------------------------------------------------------

typedef __attribute__((ext_vector_type(16))) _Float16 v16h;
typedef __attribute__((ext_vector_type(8)))  float    v8f;
typedef __attribute__((ext_vector_type(4)))  unsigned int u32x4;
typedef __attribute__((ext_vector_type(8)))  int      i32x8;
typedef __attribute__((ext_vector_type(4)))  int      i32x4;

union HFrag { v16h v; u32x4 q[2]; _Float16 h[16]; };
union H8    { u32x4 q; _Float16 h[8]; };

#define BQ   8
#define CIN  512
#define HH   64
#define WW   64
#define COUT 128
#define NPTS (BQ*HH*WW)            // 32768
#define OMS  32                    // padded 27 -> 32 channels for off/mask
#define LROW 520                   // LDS row stride in halves (512 + 4-dword TDM pad)
#define LBUF (COUT*LROW)           // halves per LDS weight buffer (66560)
#define LDS_BUF_BYTES (LBUF*2)     // 133120 B per buffer
#define LDS_TOTAL_BYTES (2*LDS_BUF_BYTES) // 266240 B dynamic LDS (two buffers)

#if __has_builtin(__builtin_amdgcn_tensor_load_to_lds)
#define HAVE_TDM 1
#else
#define HAVE_TDM 0
#endif

// ---------------------------------------------------------------------------
// 0) zero the BN-stat accumulators
__global__ void k_zero(float* s) { s[threadIdx.x] = 0.f; }

// ---------------------------------------------------------------------------
// 1) x NCHW f32 -> xh NHWC f16 (LDS tile transpose, coalesced both sides)
__global__ __launch_bounds__(256) void k_x2nhwc(const float* __restrict__ x,
                                                _Float16* __restrict__ xh) {
  __shared__ _Float16 sh[64 * 65];
  int cg = blockIdx.x & 7;
  int y  = (blockIdx.x >> 3) & 63;
  int b  = blockIdx.x >> 9;
  int c0 = cg << 6;
  for (int it = 0; it < 16; ++it) {
    int t = it * 256 + threadIdx.x;
    int xi = t & 63, ci = t >> 6;
    float v = x[((((size_t)b * CIN) + c0 + ci) * HH + y) * WW + xi];
    sh[xi * 65 + ci] = (_Float16)v;
  }
  __syncthreads();
  for (int it = 0; it < 16; ++it) {
    int t = it * 256 + threadIdx.x;
    int ci = t & 63, xi = t >> 6;
    xh[((((size_t)b * HH + y) * WW + xi) << 9) + c0 + ci] = sh[xi * 65 + ci];
  }
}

// ---------------------------------------------------------------------------
// 2) pack proj_w [O][C][3][3] f32 -> Wp [tap][O][C] f16
__global__ __launch_bounds__(256) void k_packp(const float* __restrict__ pw,
                                               _Float16* __restrict__ Wp) {
  int idx = blockIdx.x * 256 + threadIdx.x;
  if (idx >= 9 * COUT * CIN) return;
  int c = idx & 511, oc = idx >> 9, o = oc & 127, tap = oc >> 7;
  Wp[idx] = (_Float16)pw[(((size_t)o << 9) + c) * 9 + tap];
}

// pack off_w/mod_w -> Wom [tap][32][C] f16 (n: 0..17 offset, 18..26 mask, 27..31 zero)
__global__ __launch_bounds__(256) void k_packom(const float* __restrict__ ow,
                                                const float* __restrict__ mw,
                                                _Float16* __restrict__ Wom) {
  int idx = blockIdx.x * 256 + threadIdx.x;
  if (idx >= 9 * OMS * CIN) return;
  int c = idx & 511, t2 = idx >> 9, nn = t2 & 31, tap = t2 >> 5;
  float v = 0.f;
  if (nn < 18)      v = ow[(((size_t)nn << 9) + c) * 9 + tap];
  else if (nn < 27) v = mw[(((size_t)(nn - 18) << 9) + c) * 9 + tap];
  Wom[idx] = (_Float16)v;
}

// ---------------------------------------------------------------------------
// 3) offset/mask conv as WMMA GEMM: M=32768, N=32, K=9*512.
//    4 waves/block; each wave does a 16-pixel M-tile, N=0..31 (2 accumulators).
__global__ __launch_bounds__(128) void k_offmask(const _Float16* __restrict__ xh,
                                                 const _Float16* __restrict__ Wom,
                                                 const float* __restrict__ off_b,
                                                 const float* __restrict__ mod_b,
                                                 float* __restrict__ om) {
  int w = threadIdx.x >> 5, lane = threadIdx.x & 31;
  int m = lane & 15, hi = lane >> 4;
  int tile = blockIdx.x * 4 + w;
  int p0 = tile << 4;
  int b = p0 >> 12, rem = p0 & 4095, y = rem >> 6, x0 = rem & 63;
  int xi = x0 + m;
  v8f acc0 = {}, acc1 = {};

  for (int tap = 0; tap < 9; ++tap) {
    int kh = tap / 3, kw = tap % 3;
    int yy = y + kh - 1, xx = xi + kw - 1;
    bool valid = (yy >= 0) & (yy < HH) & (xx >= 0) & (xx < WW);
    const _Float16* arow =
        xh + ((((size_t)b * HH + (valid ? yy : 0)) * WW + (valid ? xx : 0)) << 9);
    const _Float16* bcol = Wom + (((size_t)tap * OMS + m) << 9) + (hi << 4);
    for (int ks = 0; ks < 16; ++ks) {
      int k0 = ks * 32;
      HFrag a;
      if (valid) {
        a.q[0] = *(const u32x4*)(arow + k0 + 16 * hi);
        a.q[1] = *(const u32x4*)(arow + k0 + 16 * hi + 8);
      } else {
        a.q[0] = (u32x4){0, 0, 0, 0};
        a.q[1] = (u32x4){0, 0, 0, 0};
      }
      // B layout: lane<16 holds K=k0..k0+15 of column n=lane; lane>=16 K=k0+16..k0+31
      v16h b0 = *(const v16h*)(bcol + k0);
      v16h b1 = *(const v16h*)(bcol + (16 << 9) + k0);
      acc0 = __builtin_amdgcn_wmma_f32_16x16x32_f16(false, a.v, false, b0,
                                                    (short)0, acc0, false, false);
      acc1 = __builtin_amdgcn_wmma_f32_16x16x32_f16(false, a.v, false, b1,
                                                    (short)0, acc1, false, false);
    }
  }
  // epilogue: +bias, clip offsets to +-16 (min(H,W)//4), mask = 2*sigmoid
  #pragma unroll
  for (int r = 0; r < 8; ++r) {
    int pp = p0 + r + 8 * hi;
    {
      int n = m;  // 0..15: all offset channels
      float v = acc0[r] + off_b[n];
      v = fminf(fmaxf(v, -16.f), 16.f);
      om[(size_t)pp * OMS + n] = v;
    }
    {
      int n = 16 + m;  // 16,17 offsets; 18..26 mask; 27..31 dead
      float v = acc1[r];
      if (n < 18) { v += off_b[n]; v = fminf(fmaxf(v, -16.f), 16.f); }
      else if (n < 27) { v += mod_b[n - 18]; v = 2.f / (1.f + expf(-v)); }
      else v = 0.f;
      om[(size_t)pp * OMS + n] = v;
    }
  }
}

// ---------------------------------------------------------------------------
// bilinear blend of 8 consecutive f16 channels from 4 neighbors (f32 math)
__device__ __forceinline__ void blend8(const _Float16* const* pj, const float* wj,
                                       int koff, _Float16* out) {
  float acc[8] = {0.f, 0.f, 0.f, 0.f, 0.f, 0.f, 0.f, 0.f};
  #pragma unroll
  for (int j = 0; j < 4; ++j) {
    H8 u; u.q = *(const u32x4*)(pj[j] + koff);
    float wv = wj[j];
    #pragma unroll
    for (int e = 0; e < 8; ++e) acc[e] += wv * (float)u.h[e];
  }
  #pragma unroll
  for (int e = 0; e < 8; ++e) out[e] = (_Float16)acc[e];
}

#if HAVE_TDM
// issue one Tensor-Data-Mover transfer: Wp tile [128 x 512] f16 -> LDS buffer,
// with 4-DWORD row padding (LROW=520 halves) so ds_load_b128 stays 16B-aligned
// and the per-lane column stride avoids LDS bank conflicts.
__device__ __forceinline__ void tdm_issue(const _Float16* gsrc, unsigned ldsOff) {
  unsigned long long ga = (unsigned long long)(size_t)gsrc;
  u32x4 g0;
  g0[0] = 1u;                                                // count = 1 descriptor
  g0[1] = ldsOff;                                            // LDS byte address
  g0[2] = (unsigned)ga;                                      // global addr lo
  g0[3] = (unsigned)((ga >> 32) & 0x1FFFFFFu) | (2u << 30);  // addr hi | type=2
  i32x8 g1;
  g1[0] = (int)((1u << 16)     // data_size = 2 bytes
              | (1u << 20)     // pad_enable
              | (7u << 22)     // pad_interval: every 256 DWORDs (one row)
              | (3u << 25));   // pad_amount: 4 DWORDs
  g1[1] = (int)(512u << 16);   // tensor_dim0 = 512
  g1[2] = (int)(128u << 16);   // tensor_dim1 = 128
  g1[3] = (int)(512u << 16);   // tile_dim0 = 512
  g1[4] = 128;                 // tile_dim1 = 128 (tile_dim2 = 0)
  g1[5] = 512;                 // tensor_dim0_stride = 512
  g1[6] = 0; g1[7] = 0;
  i32x4 g2 = (i32x4){0, 0, 0, 0};
  i32x4 g3 = (i32x4){0, 0, 0, 0};
#if defined(__clang_major__) && __clang_major__ >= 23
  i32x8 g4 = (i32x8){0, 0, 0, 0, 0, 0, 0, 0};
  __builtin_amdgcn_tensor_load_to_lds(g0, g1, g2, g3, g4, 0);
#else
  __builtin_amdgcn_tensor_load_to_lds(g0, g1, g2, g3, 0);
#endif
}
#endif

// ---------------------------------------------------------------------------
// 4) deformable core: per-tap GEMM M=32768 N=128 K=512 with on-the-fly
//    gathered A. Per-tap weights double-buffered in LDS via the TDM so the
//    next 128KB weight tile streams in while the current tap's WMMAs run.
__global__ __launch_bounds__(256) void k_deform(const _Float16* __restrict__ xh,
                                                const _Float16* __restrict__ Wp,
                                                const float* __restrict__ om,
                                                float* __restrict__ tmp) {
  extern __shared__ _Float16 shB[];   // 2 x [128][LROW] halves
  int w = threadIdx.x >> 5, lane = threadIdx.x & 31;
  int m = lane & 15, hi = lane >> 4;
  int p0 = (blockIdx.x << 7) + (w << 4);
  int p = p0 + m;
  int b = p >> 12, rem = p & 4095, y = rem >> 6, xi = rem & 63;

  v8f acc[8] = {};
  unsigned ldsOff = (unsigned)(size_t)shB;  // flat addr low 32 bits == LDS offset

#if HAVE_TDM
  if (threadIdx.x == 0) tdm_issue(Wp, ldsOff);  // prologue: tap 0 -> buffer 0
#endif

  for (int tap = 0; tap < 9; ++tap) {
    __syncthreads();  // all waves done with buffer (tap+1)&1 (used at tap-1)
#if HAVE_TDM
    if (threadIdx.x == 0) {
      if (tap < 8) {
        // start streaming the next tap's weights into the other buffer,
        // then wait only for the *older* transfer (in-order per wave).
        tdm_issue(Wp + ((size_t)(tap + 1) << 16),
                  ldsOff + (unsigned)(((tap + 1) & 1) * LDS_BUF_BYTES));
        __builtin_amdgcn_s_wait_tensorcnt(1);
      } else {
        __builtin_amdgcn_s_wait_tensorcnt(0);
      }
    }
#else
    {  // fallback: cooperative copy with padded rows into buffer tap&1
      const u32x4* src = (const u32x4*)(Wp + ((size_t)tap << 16));
      _Float16* dst = shB + (size_t)(tap & 1) * LBUF;
      for (int i = threadIdx.x; i < 4096; i += 256) {
        int row = i >> 5, col = i & 31;
        *(u32x4*)(dst + (size_t)row * LROW + (col << 3)) = src[i];
      }
    }
#endif
    __syncthreads();  // current buffer ready for all waves
    const _Float16* shBuf = shB + (size_t)(tap & 1) * LBUF;

    // ---- per-lane sample setup for this tap (mask folded into weights) ----
    int kh = tap / 3, kw = tap % 3;
    float dy = om[(size_t)p * OMS + 2 * tap];
    float dx = om[(size_t)p * OMS + 2 * tap + 1];
    float mk = om[(size_t)p * OMS + 18 + tap];
    float py = (float)y + (float)(kh - 1) + dy;
    float px = (float)xi + (float)(kw - 1) + dx;
    float y0f = floorf(py), x0f = floorf(px);
    int y0 = (int)y0f, x0 = (int)x0f;
    float wy1 = py - y0f, wx1 = px - x0f;
    float wyv[2] = {1.f - wy1, wy1}, wxv[2] = {1.f - wx1, wx1};
    const _Float16* pj[4]; float wj[4];
    #pragma unroll
    for (int jy = 0; jy < 2; ++jy)
      #pragma unroll
      for (int jx = 0; jx < 2; ++jx) {
        int yy = y0 + jy, xx = x0 + jx;
        bool v = (yy >= 0) & (yy < HH) & (xx >= 0) & (xx < WW);
        int yc = v ? yy : 0, xc = v ? xx : 0;
        pj[jy * 2 + jx] = xh + ((((size_t)b * HH + yc) * WW + xc) << 9);
        wj[jy * 2 + jx] = v ? (wyv[jy] * wxv[jx] * mk) : 0.f;
      }

    // ---- K loop: build A fragment (gather+blend), 8 WMMAs across N=128 ----
    for (int ks = 0; ks < 16; ++ks) {
      int k0 = ks * 32;
      HFrag a;
      blend8(pj, wj, k0 + 16 * hi, a.h);         // halves K = k0+16*hi .. +7
      blend8(pj, wj, k0 + 16 * hi + 8, a.h + 8); // halves K = k0+16*hi+8 .. +7
      #pragma unroll
      for (int nt = 0; nt < 8; ++nt) {
        const _Float16* bp = shBuf + (size_t)((nt << 4) + m) * LROW + k0 + (hi << 4);
        v16h bf = *(const v16h*)bp;
        acc[nt] = __builtin_amdgcn_wmma_f32_16x16x32_f16(
            false, a.v, false, bf, (short)0, acc[nt], false, false);
      }
    }
  }

  // store point-major tmp[p][cout] (coalesced); proj_b omitted (BN cancels it)
  #pragma unroll
  for (int nt = 0; nt < 8; ++nt)
    #pragma unroll
    for (int r = 0; r < 8; ++r)
      tmp[(size_t)(p0 + r + 8 * hi) * COUT + (nt << 4) + m] = acc[nt][r];
}

// ---------------------------------------------------------------------------
// 5) BN batch statistics: per-channel sum / sumsq (streaming, prefetched)
__global__ __launch_bounds__(256) void k_bnstats(const float* __restrict__ tmp,
                                                 float* __restrict__ stats) {
  int c = threadIdx.x & 127, half = threadIdx.x >> 7;
  int p0 = blockIdx.x * 128;
  float s = 0.f, s2 = 0.f;
  for (int i = half; i < 128; i += 2) {
    if (i + 8 < 128)
      __builtin_prefetch(&tmp[(size_t)(p0 + i + 8) * COUT + c], 0, 1);
    float v = tmp[(size_t)(p0 + i) * COUT + c];
    s += v; s2 += v * v;
  }
  atomicAdd(&stats[c], s);
  atomicAdd(&stats[COUT + c], s2);
}

__global__ void k_params(const float* __restrict__ stats,
                         const float* __restrict__ gamma,
                         const float* __restrict__ beta,
                         float* __restrict__ params) {
  int c = threadIdx.x;
  float n = (float)NPTS;
  float mean = stats[c] / n;
  float var = stats[COUT + c] / n - mean * mean;
  float sc = gamma[c] * rsqrtf(var + 1e-5f);
  params[c] = sc;
  params[COUT + c] = beta[c] - mean * sc;
}

// ---------------------------------------------------------------------------
// 6) normalize + exact GELU + transpose to NCHW (LDS tile, coalesced writes)
__global__ __launch_bounds__(256) void k_out(const float* __restrict__ tmp,
                                             const float* __restrict__ params,
                                             float* __restrict__ out) {
  __shared__ float sh[64 * 130];
  int b = blockIdx.x >> 6, y = blockIdx.x & 63;
  int p0 = (b << 12) + (y << 6);
  for (int it = 0; it < 32; ++it) {
    int t = it * 256 + threadIdx.x;
    int xg = t >> 7, c = t & 127;
    float v = tmp[(size_t)(p0 + xg) * COUT + c];
    v = v * params[c] + params[COUT + c];
    v = 0.5f * v * (1.f + erff(v * 0.70710678118654752f));
    sh[xg * 130 + c] = v;
  }
  __syncthreads();
  for (int it = 0; it < 32; ++it) {
    int t = it * 256 + threadIdx.x;
    int c = t >> 6, xg = t & 63;
    out[(((size_t)b * COUT + c) << 12) + (y << 6) + xg] = sh[xg * 130 + c];
  }
}

// ---------------------------------------------------------------------------
extern "C" void kernel_launch(void* const* d_in, const int* in_sizes, int n_in,
                              void* d_out, int out_size, void* d_ws, size_t ws_size,
                              hipStream_t stream) {
  const float* x      = (const float*)d_in[0];
  const float* proj_w = (const float*)d_in[1];
  // d_in[2] = proj_b : provably cancelled by training-mode BatchNorm
  const float* off_w  = (const float*)d_in[3];
  const float* off_b  = (const float*)d_in[4];
  const float* mod_w  = (const float*)d_in[5];
  const float* mod_b  = (const float*)d_in[6];
  const float* gamma  = (const float*)d_in[7];
  const float* beta   = (const float*)d_in[8];
  float* out = (float*)d_out;

  char* ws = (char*)d_ws;
  _Float16* xh     = (_Float16*)(ws);                                 // 33,554,432 B
  _Float16* Wom    = (_Float16*)(ws + 33554432);                      //    294,912 B
  _Float16* Wp     = (_Float16*)(ws + 33849344);                      //  1,179,648 B
  float*    om     = (float*)(ws + 35028992);                         //  4,194,304 B
  float*    tmp    = (float*)(ws + 39223296);                         // 16,777,216 B
  float*    stats  = (float*)(ws + 56000512);                         //      1,024 B
  float*    params = (float*)(ws + 56001536);                         //      1,024 B

  k_zero  <<<1, 256, 0, stream>>>(stats);
  k_x2nhwc<<<4096, 256, 0, stream>>>(x, xh);
  k_packp <<<(9 * COUT * CIN + 255) / 256, 256, 0, stream>>>(proj_w, Wp);
  k_packom<<<(9 * OMS * CIN + 255) / 256, 256, 0, stream>>>(off_w, mod_w, Wom);
  k_offmask<<<NPTS / 64, 128, 0, stream>>>(xh, Wom, off_b, mod_b, om);
  k_deform<<<NPTS / 128, 256, LDS_TOTAL_BYTES, stream>>>(xh, Wp, om, tmp);
  k_bnstats<<<NPTS / 128, 256, 0, stream>>>(tmp, stats);
  k_params<<<1, 128, 0, stream>>>(stats, gamma, beta, params);
  k_out   <<<BQ * HH, 256, 0, stream>>>(tmp, params, out);
}
